// ComplexMultiheadAttention_37495064494297
// MI455X (gfx1250) — matrix operand order repaired and checked by reading, not compiled, not run on hardware
//
#include <hip/hip_runtime.h>
#include <hip/hip_bf16.h>

// ---------------------------------------------------------------------------
// Types for CDNA5 WMMA (wave32). v16bf = 16 bf16 (8 VGPRs), v8f = 8 f32.
// ---------------------------------------------------------------------------
typedef __attribute__((ext_vector_type(16))) __bf16 v16bf;
typedef __attribute__((ext_vector_type(8)))  float  v8f;

#define WMMA_BF16(A, B, C) \
    __builtin_amdgcn_wmma_f32_16x16x32_bf16(false, (A), false, (B), (short)0, (C), false, false)

// Problem constants (fixed by the reference).
constexpr int BATCH = 2;
constexpr int SEQ   = 1024;
constexpr int DIM   = 1024;   // model dim == DI
constexpr int HEADS = 16;
constexpr int DH    = 64;     // head dim
constexpr int MROWS = BATCH * SEQ;  // 2048

// ---------------------------------------------------------------------------
// Fragment loaders (16-bit, wave32), using this code's contiguous k order
// A (16x32 MxK): lanes 0-15 -> M=lane; lane-half h: per lane the 16 elems are
//   K = 16h..16h+15 (contiguous, matching B).
// B (32x16 KxN): lane n = lane&15 (col), half h: elems are K = 16h..16h+15.
// C/D (16x16 f32): lane (n=lane&15, h=lane>>4), VGPR v -> row m = v+8h, col n.
// ---------------------------------------------------------------------------
__device__ __forceinline__ v16bf load_A(const __bf16* base, int ld) {
    const int lane = threadIdx.x & 31;
    const int h = lane >> 4, m = lane & 15;
    const __bf16* p = base + (size_t)m * ld + 16 * h;
    union { uint4 q[2]; v16bf v; } u;
    u.q[0] = *reinterpret_cast<const uint4*>(p);        // K = 16h .. 16h+7
    u.q[1] = *reinterpret_cast<const uint4*>(p + 8);    // K = 16h+8 .. 16h+15
    return u.v;
}

// B fragment where source is row-major [N][K] (i.e. we multiply by src^T).
__device__ __forceinline__ v16bf load_B_nt(const __bf16* base, int ld) {
    const int lane = threadIdx.x & 31;
    const int h = lane >> 4, n = lane & 15;
    const __bf16* p = base + (size_t)n * ld + 16 * h;
    union { uint4 q[2]; v16bf v; } u;
    u.q[0] = *reinterpret_cast<const uint4*>(p);        // K = 16h .. 16h+7
    u.q[1] = *reinterpret_cast<const uint4*>(p + 8);    // K = 16h+8 .. 16h+15
    return u.v;
}

// A fragment where source is row-major [K][M] (transposed A, e.g. V^T from LDS).
__device__ __forceinline__ v16bf load_A_t(const __bf16* base, int ld) {
    const int lane = threadIdx.x & 31;
    const int h = lane >> 4, m = lane & 15;
    v16bf r;
#pragma unroll
    for (int i = 0; i < 16; ++i) {
        const int v = i >> 1, pbit = i & 1;
        const int k = 16 * h + i; (void)v; (void)pbit;
        r[i] = base[(size_t)k * ld + m];
    }
    return r;
}

// B fragment from a [K][N] row-major buffer (used for P^T staged in LDS).
__device__ __forceinline__ v16bf load_B_kn(const __bf16* base, int ld) {
    const int lane = threadIdx.x & 31;
    const int h = lane >> 4, n = lane & 15;
    v16bf r;
#pragma unroll
    for (int i = 0; i < 16; ++i) r[i] = base[(size_t)(16 * h + i) * ld + n];
    return r;
}

// ---------------------------------------------------------------------------
// Async global -> LDS 16-byte copy (CDNA5 async-DMA path, ASYNCcnt tracked).
// LDS byte offset = low 32 bits of the generic pointer (LDS aperture rule).
// ---------------------------------------------------------------------------
__device__ __forceinline__ void stage16_async(const __bf16* g, __bf16* l) {
    const unsigned lds_off = (unsigned)(unsigned long long)(void*)l;
    const unsigned long long gaddr = (unsigned long long)g;
    asm volatile("global_load_async_to_lds_b128 %0, %1, off"
                 :: "v"(lds_off), "v"(gaddr) : "memory");
}
__device__ __forceinline__ void wait_async0() {
    asm volatile("s_wait_asynccnt 0x0" ::: "memory");
}

// ---------------------------------------------------------------------------
// fp32 -> bf16 cast
// ---------------------------------------------------------------------------
__global__ void cast_f32_to_bf16(const float* __restrict__ src,
                                 __bf16* __restrict__ dst, size_t n) {
    size_t i = (size_t)blockIdx.x * blockDim.x + threadIdx.x;
    if (i < n) dst[i] = (__bf16)src[i];
}

// ---------------------------------------------------------------------------
// Complex GEMM:  Yr = Xr@Wr^T - Xi@Wi^T ;  Yi = Xr@Wi^T + Xi@Wr^T
// blockIdx.z selects real/imag output. One wave per 16x16 tile, 4 waves/block.
// ---------------------------------------------------------------------------
template <bool OUT_BF16>
__global__ __launch_bounds__(128) void cgemm_kernel(
    const __bf16* __restrict__ Xr, const __bf16* __restrict__ Xi,
    const __bf16* __restrict__ Wr, const __bf16* __restrict__ Wi,
    void* __restrict__ Yr, void* __restrict__ Yi,
    int Nout, int K, float out_scale)
{
    const int wave = threadIdx.x >> 5;
    const int m0 = blockIdx.x * 16;
    const int n0 = blockIdx.y * 64 + wave * 16;
    const int z  = blockIdx.z;

    const __bf16* W1 = z ? Wi : Wr;
    const __bf16* W2 = z ? Wr : Wi;
    const float sign = z ? 1.0f : -1.0f;

    v8f acc1 = {}, acc2 = {};
    for (int k0 = 0; k0 < K; k0 += 32) {
        v16bf a1 = load_A(Xr + (size_t)m0 * K + k0, K);
        v16bf b1 = load_B_nt(W1 + (size_t)n0 * K + k0, K);
        acc1 = WMMA_BF16(a1, b1, acc1);
        v16bf a2 = load_A(Xi + (size_t)m0 * K + k0, K);
        v16bf b2 = load_B_nt(W2 + (size_t)n0 * K + k0, K);
        acc2 = WMMA_BF16(a2, b2, acc2);
    }

    const int lane = threadIdx.x & 31;
    const int h = lane >> 4, n = lane & 15;
    void* Y = z ? Yi : Yr;
#pragma unroll
    for (int v = 0; v < 8; ++v) {
        const float val = (acc1[v] + sign * acc2[v]) * out_scale;
        const size_t idx = (size_t)(m0 + v + 8 * h) * Nout + (n0 + n);
        if constexpr (OUT_BF16) ((__bf16*)Y)[idx] = (__bf16)val;
        else                    ((float*)Y)[idx]  = val;
    }
}

// ---------------------------------------------------------------------------
// Flash attention over the 4 sign-groups. Block = 128 threads = 4 waves,
// wave g handles group g for one (b, head, 16-query tile).
//   g0:(qr,kr,vr) g1:(qr,ki,vi) g2:(qi,kr,vr) g3:(qi,ki,vi)
//   o_r = o0-o1-o2-o3 ; o_i = o0+o1+o2-o3   (signs applied via LDS atomics)
// K/V tiles (both r and i variants) are staged cooperatively into LDS via
// global_load_async_to_lds_b128 (double-buffered, 1 barrier / 32-key step);
// scores computed transposed (S^T = K@Q^T) so each query column sits in a
// lane pair -> softmax reductions are a single shfl_xor(16).
// ---------------------------------------------------------------------------
__global__ __launch_bounds__(128) void attn_kernel(
    const __bf16* __restrict__ q_r, const __bf16* __restrict__ q_i,
    const __bf16* __restrict__ kv_r, const __bf16* __restrict__ kv_i,
    __bf16* __restrict__ o_r, __bf16* __restrict__ o_i)
{
    constexpr int LDQ = DIM, LDKV = 2 * DIM;
    __shared__ __bf16 kvtile[2][4][32 * 64]; // [buf][Kr,Ki,Vr,Vi][32 x 64], 32 KB
    __shared__ __bf16 pbuf[4][32 * 16];      // per-wave P^T tile (32 keys x 16 q)
    __shared__ float obr[16 * 64];           // combined o_r (16 q x 64 d)
    __shared__ float obi[16 * 64];

    const int tid = threadIdx.x;
    for (int i = tid; i < 16 * 64; i += 128) { obr[i] = 0.0f; obi[i] = 0.0f; }

    const int g = tid >> 5;
    const int lane = tid & 31, hh = lane >> 4, n = lane & 15;
    const int qt = blockIdx.x;               // query tile (16 rows)
    const int b  = blockIdx.y >> 4;
    const int hd = blockIdx.y & 15;          // head

    const __bf16* qsel = (g >= 2) ? q_i : q_r;
    const __bf16* qptr = qsel + (size_t)(b * SEQ + qt * 16) * LDQ + hd * DH;
    const __bf16* kv_row0_r = kv_r + (size_t)(b * SEQ) * LDKV + hd * DH;
    const __bf16* kv_row0_i = kv_i + (size_t)(b * SEQ) * LDKV + hd * DH;

    // Q^T fragments (B operands), d = 0..31 and 32..63. Scale 1/8 is baked into q.
    const v16bf bq0 = load_B_nt(qptr, LDQ);
    const v16bf bq1 = load_B_nt(qptr + 32, LDQ);

    v8f acc[4]; acc[0] = v8f{}; acc[1] = v8f{}; acc[2] = v8f{}; acc[3] = v8f{};
    float m_run = -__builtin_huge_valf();
    float l_run = 0.0f;
    __bf16* pw = pbuf[g];

    int p = 0;
    for (int j = 0; j < SEQ; j += 32) {
        // ---- cooperative async staging of Kr,Ki,Vr,Vi tiles (16 KB) ----
        // 1024 chunks of 16 B; 8 per thread; fully coalesced per tile row.
#pragma unroll
        for (int it = 0; it < 8; ++it) {
            const int c    = tid + it * 128;
            const int tile = c >> 8;              // 0:Kr 1:Ki 2:Vr 3:Vi
            const int cc   = c & 255;
            const int row  = cc >> 3;
            const int col  = (cc & 7) * 8;        // bf16 elements
            const __bf16* src = ((tile & 1) ? kv_row0_i : kv_row0_r)
                              + (size_t)(j + row) * LDKV + (tile >> 1) * DIM + col;
            stage16_async(src, &kvtile[p][tile][row * 64 + col]);
        }
        wait_async0();
        __syncthreads();

        const __bf16* Ksh = kvtile[p][(g & 1)];
        const __bf16* Vsh = kvtile[p][2 + (g & 1)];

        // S^T (32 keys x 16 queries) = K_j @ Q^T, two 16x16 C tiles (from LDS)
        v8f s0 = {}, s1 = {};
        s0 = WMMA_BF16(load_A(Ksh, DH),           bq0, s0);
        s0 = WMMA_BF16(load_A(Ksh + 32, DH),      bq1, s0);
        s1 = WMMA_BF16(load_A(Ksh + 16 * DH, DH), bq0, s1);
        s1 = WMMA_BF16(load_A(Ksh + 16 * DH + 32, DH), bq1, s1);

        // --- online softmax per query column (lane pair lane, lane^16) ---
        float tmax = s0[0];
#pragma unroll
        for (int v = 0; v < 8; ++v) { tmax = fmaxf(tmax, s0[v]); tmax = fmaxf(tmax, s1[v]); }
        tmax = fmaxf(tmax, __shfl_xor(tmax, 16, 32));
        const float m_new = fmaxf(m_run, tmax);
        const float alpha = __expf(m_run - m_new);
        float p0[8], p1[8], psum = 0.0f;
#pragma unroll
        for (int v = 0; v < 8; ++v) {
            p0[v] = __expf(s0[v] - m_new);
            p1[v] = __expf(s1[v] - m_new);
            psum += p0[v] + p1[v];
        }
        psum += __shfl_xor(psum, 16, 32);
        l_run = l_run * alpha + psum;
        m_run = m_new;
#pragma unroll
        for (int t = 0; t < 4; ++t)
#pragma unroll
            for (int v = 0; v < 8; ++v) acc[t][v] *= alpha;

        // Stage P^T (bf16) into per-wave LDS slice in [key][query] order;
        // same-wave DS ops are in-order so no cross-wave barrier needed.
#pragma unroll
        for (int v = 0; v < 8; ++v) {
            pw[(v + 8 * hh) * 16 + n]      = (__bf16)p0[v];   // keys 0..15
            pw[(16 + v + 8 * hh) * 16 + n] = (__bf16)p1[v];   // keys 16..31
        }
        asm volatile("" ::: "memory");
        const v16bf bp = load_B_kn(pw, 16);
        asm volatile("" ::: "memory");

        // O^T (64 x 16q) += V_j^T (64 x 32keys) @ P^T, four d-blocks of 16.
        // V^T gather comes from LDS (cheap) instead of strided global loads.
#pragma unroll
        for (int t = 0; t < 4; ++t)
            acc[t] = WMMA_BF16(load_A_t(Vsh + t * 16, DH), bp, acc[t]);

        p ^= 1;   // double buffer: next staging can overlap this compute
    }

    // Normalize and combine the 4 groups with sign vectors via LDS atomics.
    const float inv_l = 1.0f / l_run;
    const float sr = (g == 0) ? 1.0f : -1.0f;
    const float si = (g == 3) ? -1.0f : 1.0f;
#pragma unroll
    for (int t = 0; t < 4; ++t)
#pragma unroll
        for (int v = 0; v < 8; ++v) {
            const int d = t * 16 + v + 8 * hh;       // acc layout: row d, col n (query)
            const float val = acc[t][v] * inv_l;
            atomicAdd(&obr[n * 64 + d], sr * val);
            atomicAdd(&obi[n * 64 + d], si * val);
        }
    __syncthreads();

    // Store merged-heads o_r / o_i as bf16: [B*SEQ, HEADS*DH] row-major.
    for (int i = tid; i < 16 * 64; i += 128) {
        const int q = i >> 6, d = i & 63;
        const size_t idx = (size_t)(b * SEQ + qt * 16 + q) * DIM + hd * DH + d;
        o_r[idx] = (__bf16)obr[i];
        o_i[idx] = (__bf16)obi[i];
    }
}

// ---------------------------------------------------------------------------
// Host launch
// ---------------------------------------------------------------------------
extern "C" void kernel_launch(void* const* d_in, const int* in_sizes, int n_in,
                              void* d_out, int out_size, void* d_ws, size_t ws_size,
                              hipStream_t stream) {
    (void)in_sizes; (void)n_in; (void)out_size; (void)ws_size;

    const float* x_r    = (const float*)d_in[0];
    const float* x_i    = (const float*)d_in[1];
    const float* wq_r   = (const float*)d_in[2];
    const float* wq_i   = (const float*)d_in[3];
    const float* wkv_r  = (const float*)d_in[4];
    const float* wkv_i  = (const float*)d_in[5];
    const float* wout_r = (const float*)d_in[6];
    const float* wout_i = (const float*)d_in[7];
    float* out = (float*)d_out;

    // Workspace layout (bf16 elements). Total 28M elems = 56 MB.
    constexpr size_t M1 = 1024ull * 1024;
    __bf16* w = (__bf16*)d_ws;
    __bf16* xr_b    = w + 0 * M1;    // 2M
    __bf16* xi_b    = w + 2 * M1;    // 2M
    __bf16* wqr_b   = w + 4 * M1;    // 1M
    __bf16* wqi_b   = w + 5 * M1;    // 1M
    __bf16* wkvr_b  = w + 6 * M1;    // 2M
    __bf16* wkvi_b  = w + 8 * M1;    // 2M
    __bf16* woutr_b = w + 10 * M1;   // 1M
    __bf16* wouti_b = w + 11 * M1;   // 1M
    __bf16* q_r     = w + 12 * M1;   // 2M  (pre-scaled by 1/8)
    __bf16* q_i     = w + 14 * M1;   // 2M
    __bf16* kv_r    = w + 16 * M1;   // 4M
    __bf16* kv_i    = w + 20 * M1;   // 4M
    __bf16* o_r     = w + 24 * M1;   // 2M
    __bf16* o_i     = w + 26 * M1;   // 2M

    auto cast = [&](const float* s, __bf16* d, size_t n) {
        cast_f32_to_bf16<<<dim3((unsigned)((n + 255) / 256)), dim3(256), 0, stream>>>(s, d, n);
    };
    cast(x_r,    xr_b,    2 * M1);
    cast(x_i,    xi_b,    2 * M1);
    cast(wq_r,   wqr_b,   1 * M1);
    cast(wq_i,   wqi_b,   1 * M1);
    cast(wkv_r,  wkvr_b,  2 * M1);
    cast(wkv_i,  wkvi_b,  2 * M1);
    cast(wout_r, woutr_b, 1 * M1);
    cast(wout_i, wouti_b, 1 * M1);

    // Q projection (scale 1/sqrt(64) folded into q), bf16 output.
    cgemm_kernel<true><<<dim3(MROWS / 16, DIM / 64, 2), dim3(128), 0, stream>>>(
        xr_b, xi_b, wqr_b, wqi_b, q_r, q_i, DIM, DIM, 0.125f);

    // KV projection, bf16 output [2048 x 2048] (cols 0..1023 = K, 1024..2047 = V).
    cgemm_kernel<true><<<dim3(MROWS / 16, (2 * DIM) / 64, 2), dim3(128), 0, stream>>>(
        xr_b, xi_b, wkvr_b, wkvi_b, kv_r, kv_i, 2 * DIM, DIM, 1.0f);

    // Attention: grid = (query tiles, B*HEADS), 4 waves = 4 sign-groups.
    attn_kernel<<<dim3(SEQ / 16, BATCH * HEADS), dim3(128), 0, stream>>>(
        q_r, q_i, kv_r, kv_i, o_r, o_i);

    // Output projection -> fp32 d_out: y_r at [0, 2M), y_i at [2M, 4M).
    cgemm_kernel<false><<<dim3(MROWS / 16, DIM / 64, 2), dim3(128), 0, stream>>>(
        o_r, o_i, woutr_b, wouti_b, out, out + 2 * M1, DIM, DIM, 1.0f);
}
